// AdvancedMixConsole_63806034149970
// MI455X (gfx1250) — matrix operand order repaired and checked by reading, not compiled, not run on hardware
//
#include <hip/hip_runtime.h>
#include <math.h>

#define T_LEN   262144
#define NFFT    524288
#define NROW    1024
#define NCOL    512
#define NSAMP   65536
#define NTAPS   1023
#define WNLEN   66558
#define FPAD    1088
#define SRATE   48000.0
#define PI_D    3.14159265358979323846
#define TWO_PI_F 6.28318530718f

typedef _Float16     v16h __attribute__((ext_vector_type(16)));
typedef float        v8f  __attribute__((ext_vector_type(8)));
typedef unsigned int v4u  __attribute__((ext_vector_type(4)));
typedef int          v8i  __attribute__((ext_vector_type(8)));
typedef int          v4i  __attribute__((ext_vector_type(4)));

struct SigP {
  float gain;
  float bq[6][5];              // b0 b1 b2 a1 a2 (normalized by a0)
  float thr, ratio, knee, alphaA, makeup;
  float cosT, sinT, sendLin;
};

__device__ __forceinline__ float2 cxmul(float2 a, float2 b) {
  return make_float2(a.x*b.x - a.y*b.y, a.x*b.y + a.y*b.x);
}

__device__ __forceinline__ double dsinc(double x) {
  if (x == 0.0) return 1.0;
  double px = PI_D * x;
  return sin(px) / px;
}

// ---------------------------------------------------------------------------
// Parameter precompute: 32 track entries + 8 master entries (b*2+ch)
// ---------------------------------------------------------------------------
__global__ void k_params(SigP* __restrict__ P, const float* __restrict__ tp,
                         const float* __restrict__ mbp) {
  int tid = blockIdx.x * blockDim.x + threadIdx.x;
  if (tid >= 40) return;
  const float eqlo[18] = {-12.f,20.f,0.1f, -12.f,80.f,0.1f, -12.f,2000.f,0.1f,
                          -12.f,8000.f,0.1f, -12.f,12000.f,0.1f, -12.f,6000.f,0.1f};
  const float eqhi[18] = {12.f,2000.f,5.f, 12.f,2000.f,5.f, 12.f,8000.f,5.f,
                          12.f,12000.f,5.f, 12.f,23000.f,5.f, 12.f,23000.f,5.f};
  const float clo[6] = {-60.f,1.f,1.f,1.f,3.f,0.f};
  const float chi[6] = {0.f,10.f,1000.f,1000.f,24.f,24.f};
  float eqp[18], cp[6];
  SigP sp;
  if (tid < 32) {
    const float* p = tp + tid * 27;
    sp.gain = powf(10.f, (p[0]*24.f - 12.f) * 0.05f);
    for (int i = 0; i < 18; ++i) eqp[i] = p[1+i]*(eqhi[i]-eqlo[i]) + eqlo[i];
    for (int i = 0; i < 6;  ++i) cp[i]  = p[19+i]*(chi[i]-clo[i]) + clo[i];
    float th = p[25] * 1.57079632679f;
    sp.cosT = cosf(th); sp.sinT = sinf(th);
    sp.sendLin = powf(10.f, (p[26]*92.f - 80.f) * 0.05f);
  } else {
    const float* p = mbp + ((tid - 32) >> 1) * 24;
    sp.gain = 1.f;
    for (int i = 0; i < 18; ++i) eqp[i] = p[i]*(eqhi[i]-eqlo[i]) + eqlo[i];
    for (int i = 0; i < 6;  ++i) cp[i]  = p[18+i]*(chi[i]-clo[i]) + clo[i];
    sp.cosT = 0.f; sp.sinT = 0.f; sp.sendLin = 0.f;
  }
  for (int s = 0; s < 6; ++s) {
    double g = eqp[3*s], fc = eqp[3*s+1], q = eqp[3*s+2];
    double A  = pow(10.0, g/40.0);
    double w0 = 2.0*PI_D*fc/SRATE;
    double al = sin(w0)/(2.0*q);
    double c  = cos(w0);
    double sA = sqrt(A);
    double b0,b1,b2,a0,a1,a2;
    if (s == 0) {        // low shelf
      b0 = A*((A+1.0) - (A-1.0)*c + 2.0*sA*al);
      b1 = 2.0*A*((A-1.0) - (A+1.0)*c);
      b2 = A*((A+1.0) - (A-1.0)*c - 2.0*sA*al);
      a0 = (A+1.0) + (A-1.0)*c + 2.0*sA*al;
      a1 = -2.0*((A-1.0) + (A+1.0)*c);
      a2 = (A+1.0) + (A-1.0)*c - 2.0*sA*al;
    } else if (s == 5) { // high shelf
      b0 = A*((A+1.0) + (A-1.0)*c + 2.0*sA*al);
      b1 = -2.0*A*((A-1.0) + (A+1.0)*c);
      b2 = A*((A+1.0) + (A-1.0)*c - 2.0*sA*al);
      a0 = (A+1.0) - (A-1.0)*c + 2.0*sA*al;
      a1 = 2.0*((A-1.0) - (A+1.0)*c);
      a2 = (A+1.0) - (A-1.0)*c - 2.0*sA*al;
    } else {             // peaking
      b0 = 1.0 + al*A; b1 = -2.0*c; b2 = 1.0 - al*A;
      a0 = 1.0 + al/A; a1 = -2.0*c; a2 = 1.0 - al/A;
    }
    double ia = 1.0/a0;
    sp.bq[s][0] = (float)(b0*ia); sp.bq[s][1] = (float)(b1*ia); sp.bq[s][2] = (float)(b2*ia);
    sp.bq[s][3] = (float)(a1*ia); sp.bq[s][4] = (float)(a2*ia);
  }
  sp.thr = cp[0]; sp.ratio = cp[1]; sp.knee = cp[4]; sp.makeup = cp[5];
  sp.alphaA = (float)exp(-log(9.0) / (SRATE * (double)cp[2] / 1000.0));
  P[tid] = sp;
}

// ---------------------------------------------------------------------------
// On-device firwin filterbank design (12 bands, 1023 taps, reversed, f16,
// padded by 16 zeros front / 49 back for the Toeplitz-WMMA indexing)
// ---------------------------------------------------------------------------
__global__ void k_make_filters(_Float16* __restrict__ filtH) {
  __shared__ double red[256];
  int band = blockIdx.x;
  int t = threadIdx.x;
  const double BANDS[10] = {31.5,63.0,125.0,250.0,500.0,1000.0,2000.0,4000.0,8000.0,16000.0};
  double l, r, sf;
  if (band == 0)       { l = 0.0;  r = 12.0/24000.0; sf = 0.0; }
  else if (band == 11) { l = 0.75; r = 1.0;          sf = 1.0; }
  else {
    double fc = BANDS[band-1];
    double fmin = fc / sqrt(2.0);
    double fmax = fc * sqrt(2.0);
    if (fmax > 23976.0) fmax = 23976.0;
    l = fmin/24000.0; r = fmax/24000.0; sf = 0.5*(l+r);
  }
  double hv[4];
  double part = 0.0;
  for (int u = 0; u < 4; ++u) {
    int i = t + u*256;
    if (i < NTAPS) {
      double m = (double)i - 511.0;
      double hh = r*dsinc(r*m) - l*dsinc(l*m);
      hh *= 0.54 - 0.46*cos(2.0*PI_D*(double)i/1022.0);   // hamming
      hv[u] = hh;
      part += hh * cos(PI_D*m*sf);
    } else hv[u] = 0.0;
  }
  red[t] = part; __syncthreads();
  for (int s = 128; s > 0; s >>= 1) { if (t < s) red[t] += red[t+s]; __syncthreads(); }
  double norm = red[0];
  for (int u = 0; u < 4; ++u) {
    int i = t + u*256;
    if (i < NTAPS)
      filtH[band*FPAD + 16 + (1022 - i)] = (_Float16)(hv[u]/norm);  // reversed
  }
  if (t < 16) filtH[band*FPAD + t] = (_Float16)0.0f;
  if (t < FPAD - (16 + NTAPS)) filtH[band*FPAD + 16 + NTAPS + t] = (_Float16)0.0f;
}

// ---------------------------------------------------------------------------
// Four-step FFT, N = 524288 = 1024 (rows) x 512 (cols).
// Forward: cols(dir=-1, twiddle) then rows(dir=-1). Spectrum stored [k2][k1],
//   true bin k = k2 + 512*k1 at position p = k2*1024 + k1.
// Inverse: rows(dir=+1, twiddle) then cols(dir=+1, scale=1/N) -> natural order.
// Both kernels are in-place (workgroup owns its row / its 4 columns).
// ---------------------------------------------------------------------------
__global__ void k_fft_rows(float2* __restrict__ buf, float dir, int doTw, float scale) {
  __shared__ float2 As[1024];
  __shared__ float2 Bs[1024];
  int row = blockIdx.x;
  size_t base = (size_t)blockIdx.y * NFFT + (size_t)row * NROW;
  int t = threadIdx.x;
  for (int i = t; i < 1024; i += 256) As[i] = buf[base + i];
  __syncthreads();
  float2* src = As; float2* dst = Bs;
  for (int Ns = 1; Ns < 1024; Ns <<= 1) {         // 10 stages (even -> ends in As)
    for (int j = t; j < 512; j += 256) {
      int grp = j / Ns, pos = j - grp*Ns;
      int ob = grp*(Ns << 1) + pos;
      float ang = dir * 3.14159265f * (float)pos / (float)Ns;
      float sn, cs; __sincosf(ang, &sn, &cs);
      float2 a = src[j], b = src[j + 512];
      float2 wb = make_float2(cs*b.x - sn*b.y, cs*b.y + sn*b.x);
      dst[ob]      = make_float2(a.x + wb.x, a.y + wb.y);
      dst[ob + Ns] = make_float2(a.x - wb.x, a.y - wb.y);
    }
    __syncthreads();
    float2* tmp = src; src = dst; dst = tmp;
  }
  for (int i = t; i < 1024; i += 256) {
    float2 v = src[i];
    if (doTw) {
      float ang = dir * TWO_PI_F * (float)(i * row) / (float)NFFT;
      float sn, cs; __sincosf(ang, &sn, &cs);
      v = make_float2(cs*v.x - sn*v.y, cs*v.y + sn*v.x);
    }
    buf[base + i] = make_float2(v.x*scale, v.y*scale);
  }
}

__global__ void k_fft_cols(float2* __restrict__ buf, float dir, int doTw, float scale) {
  __shared__ float2 As[4*512];
  __shared__ float2 Bs[4*512];
  int c0 = blockIdx.x * 4;
  size_t sbase = (size_t)blockIdx.y * NFFT;
  int t = threadIdx.x;
  for (int l = t; l < 2048; l += 256) {
    int col = l & 3, j = l >> 2;
    As[col*512 + j] = buf[sbase + (size_t)(c0 + col) + (size_t)j * NROW];
  }
  __syncthreads();
  float2* src = As; float2* dst = Bs;
  for (int Ns = 1; Ns < 512; Ns <<= 1) {          // 9 stages
    for (int l = t; l < 1024; l += 256) {
      int col = l >> 8, j = l & 255;
      int grp = j / Ns, pos = j - grp*Ns;
      int ob = grp*(Ns << 1) + pos;
      float ang = dir * 3.14159265f * (float)pos / (float)Ns;
      float sn, cs; __sincosf(ang, &sn, &cs);
      float2 a = src[col*512 + j], b = src[col*512 + j + 256];
      float2 wb = make_float2(cs*b.x - sn*b.y, cs*b.y + sn*b.x);
      dst[col*512 + ob]      = make_float2(a.x + wb.x, a.y + wb.y);
      dst[col*512 + ob + Ns] = make_float2(a.x - wb.x, a.y - wb.y);
    }
    __syncthreads();
    float2* tmp = src; src = dst; dst = tmp;
  }
  for (int l = t; l < 2048; l += 256) {
    int col = l & 3, j = l >> 2;
    float2 v = src[col*512 + j];
    if (doTw) {
      float ang = dir * TWO_PI_F * (float)((c0 + col) * j) / (float)NFFT;
      float sn, cs; __sincosf(ang, &sn, &cs);
      v = make_float2(cs*v.x - sn*v.y, cs*v.y + sn*v.x);
    }
    buf[sbase + (size_t)(c0 + col) + (size_t)j * NROW] = make_float2(v.x*scale, v.y*scale);
  }
}

// ---------------------------------------------------------------------------
// Elementwise stages
// ---------------------------------------------------------------------------
__global__ void k_pack_tracks(float2* __restrict__ buf, const float* __restrict__ tracks,
                              const SigP* __restrict__ P, int batch) {
  int sig = blockIdx.y;
  int i = blockIdx.x*256 + threadIdx.x;
  float g = P[batch*8 + sig].gain;
  const float* src = tracks + ((size_t)batch*8 + sig)*T_LEN + i;
  __builtin_prefetch(src + 8192, 0, 1);           // global_prefetch (speculative)
  float v = (i < T_LEN) ? (*src) * g : 0.f;
  buf[(size_t)sig*NFFT + i] = make_float2(v, 0.f);
}

__global__ void k_pack_real(float2* __restrict__ buf, const float* __restrict__ src, int L) {
  int sig = blockIdx.y;
  int i = blockIdx.x*256 + threadIdx.x;
  float v = (i < L) ? src[(size_t)sig*L + i] : 0.f;
  buf[(size_t)sig*NFFT + i] = make_float2(v, 0.f);
}

__global__ void k_eq_mul(float2* __restrict__ buf, const SigP* __restrict__ P, int sbase) {
  int sig = blockIdx.y;
  int p = blockIdx.x*256 + threadIdx.x;
  int k2 = p >> 10, k1 = p & 1023;
  int k = k2 + 512*k1;                       // true frequency bin
  float wang = TWO_PI_F * (float)k / (float)NFFT;
  float sn, cs; __sincosf(wang, &sn, &cs);
  float zr = cs, zi = -sn;                   // z = e^{-i w}
  float z2r = zr*zr - zi*zi, z2i = 2.f*zr*zi;
  const SigP sp = P[sbase + sig];
  float Hr = 1.f, Hi = 0.f;
#pragma unroll
  for (int s = 0; s < 6; ++s) {
    float b0 = sp.bq[s][0], b1 = sp.bq[s][1], b2 = sp.bq[s][2];
    float a1 = sp.bq[s][3], a2 = sp.bq[s][4];
    float nr = b0 + b1*zr + b2*z2r;
    float ni =      b1*zi + b2*z2i;
    float dr = 1.f + a1*zr + a2*z2r;
    float di =       a1*zi + a2*z2i;
    float id = 1.f / (dr*dr + di*di);
    float tr = (nr*dr + ni*di)*id;
    float ti = (ni*dr - nr*di)*id;
    float hr = Hr*tr - Hi*ti;
    Hi = Hr*ti + Hi*tr; Hr = hr;
  }
  size_t idx = (size_t)sig*NFFT + p;
  float2 X = buf[idx];
  buf[idx] = make_float2(X.x*Hr - X.y*Hi, X.x*Hi + X.y*Hr);
}

__global__ void k_h1p_mul(float2* __restrict__ buf, const SigP* __restrict__ P, int sbase) {
  int sig = blockIdx.y;
  int p = blockIdx.x*256 + threadIdx.x;
  int k2 = p >> 10, k1 = p & 1023;
  int k = k2 + 512*k1;
  float wang = TWO_PI_F * (float)k / (float)NFFT;
  float sn, cs; __sincosf(wang, &sn, &cs);
  float zr = cs, zi = -sn;
  float a = P[sbase + sig].alphaA;
  float dr = 1.f - a*zr;
  float dim = -a*zi;
  float id = (1.f - a) / (dr*dr + dim*dim);
  float Hr = dr*id, Hi = -dim*id;
  size_t idx = (size_t)sig*NFFT + p;
  float2 X = buf[idx];
  buf[idx] = make_float2(X.x*Hr - X.y*Hi, X.x*Hi + X.y*Hr);
}

__global__ void k_comp_gc(float2* __restrict__ buf, float* __restrict__ xG,
                          const SigP* __restrict__ P, int sbase) {
  int sig = blockIdx.y;
  int i = blockIdx.x*256 + threadIdx.x;
  size_t idx = (size_t)sig*NFFT + i;
  if (i < T_LEN) {
    float thr = P[sbase+sig].thr;
    float R   = P[sbase+sig].ratio;
    float W   = P[sbase+sig].knee;
    float x = buf[idx].x;
    xG[(size_t)sig*T_LEN + i] = x;
    float ax = fmaxf(fabsf(x), 1e-8f);
    float xdb = 20.f * log10f(ax);
    float e = xdb - thr + 0.5f*W;
    float knee_out = xdb + (1.f/R - 1.f)*e*e/(2.f*W);
    float above = thr + (xdb - thr)/R;
    float xsc = (xdb < thr - 0.5f*W) ? xdb : ((xdb <= thr + 0.5f*W) ? knee_out : above);
    buf[idx] = make_float2(xsc - xdb, 0.f);
  } else {
    buf[idx] = make_float2(0.f, 0.f);
  }
}

__global__ void k_comp_mix(const float2* __restrict__ buf, const float* __restrict__ xG,
                           float* __restrict__ master, float* __restrict__ bus,
                           const SigP* __restrict__ P, int batch) {
  int sig = blockIdx.y;
  int i = blockIdx.x*256 + threadIdx.x;
  const SigP* sp = &P[batch*8 + sig];
  float gs = buf[(size_t)sig*NFFT + i].x + sp->makeup;
  float y = xG[(size_t)sig*T_LEN + i] * __expf(gs * 0.11512925465f);  // 10^(gs/20)
  float yl = y * sp->cosT, yr = y * sp->sinT;
  size_t o0 = (size_t)(batch*2 + 0)*T_LEN + i;
  size_t o1 = (size_t)(batch*2 + 1)*T_LEN + i;
  atomicAdd(&master[o0], yl);
  atomicAdd(&master[o1], yr);
  float s = sp->sendLin;
  atomicAdd(&bus[o0], yl*s);
  atomicAdd(&bus[o1], yr*s);
}

__global__ void k_cmul(float2* __restrict__ a, const float2* __restrict__ b) {
  size_t i = (size_t)blockIdx.y*NFFT + blockIdx.x*256 + threadIdx.x;
  a[i] = cxmul(a[i], b[i]);
}

__global__ void k_wet_mix(float* __restrict__ master, const float* __restrict__ bus,
                          const float2* __restrict__ wet, const float* __restrict__ fxp) {
  int sig = blockIdx.y;
  int i = blockIdx.x*256 + threadIdx.x;
  int b = sig >> 1;
  float m = fxp[b*25 + 24];
  size_t o = (size_t)sig*T_LEN + i;
  master[o] += (1.f - m)*bus[o] + m*wet[(size_t)sig*NFFT + i].x;
}

__global__ void k_final(const float2* __restrict__ buf, const float* __restrict__ xG,
                        const SigP* __restrict__ P, float* __restrict__ out) {
  int sig = blockIdx.y;
  int i = blockIdx.x*256 + threadIdx.x;
  float gs = buf[(size_t)sig*NFFT + i].x + P[32 + sig].makeup;
  out[(size_t)sig*T_LEN + i] = xG[(size_t)sig*T_LEN + i] * __expf(gs * 0.11512925465f);
}

// ---------------------------------------------------------------------------
// WMMA FIR filterbank + fused IR synthesis.
// One wave per block. C[m][n] = sum_j A[m][j]*B[j][n] with
//   A[m][j] = x[b, t0 + 16*blk + j]  (m = 8*blk + b)
//   B[j][n] = hrev[j - n]            (Toeplitz filter in LDS, offset +16)
// => C[m][n] = out[b, t0 + 16*blk + n]. f16 in, f32 accumulate.
// Data movement uses the CDNA5 paths:
//   - filter row    : TDM tensor_load_to_lds (TENSORcnt)
//   - sample slab   : global_load_async_to_lds_b64 (ASYNCcnt)
// Fused epilogue: ir[batch,ch,t] += C * gain * exp(-d*t/65535) / 12
// ---------------------------------------------------------------------------
__global__ void __launch_bounds__(32)
k_fir_wmma(const float* __restrict__ wn, const _Float16* __restrict__ filtH,
           const float* __restrict__ fxp, float* __restrict__ ir) {
  __shared__ __align__(16) float    xf[8*FPAD];   // f32 slab landed by async copy
  __shared__ __align__(16) _Float16 xs[8*FPAD];   // f16 A-operand slab
  __shared__ __align__(16) _Float16 hs[FPAD];     // f16 Toeplitz filter row
  int lane = threadIdx.x;
  int band = blockIdx.y;
  int t0 = blockIdx.x * 32;

  // ---- TDM: DMA the 2176B filter row into LDS (1D tile descriptor) ----
  {
    unsigned ldsA = (unsigned)(size_t)(&hs[0]);               // low 32b = LDS offset
    unsigned long long ga = (unsigned long long)(size_t)(filtH + band*FPAD);
    v4u g0 = { 1u,                                            // count=1, user mode
               ldsA,                                          // lds_addr
               (unsigned)(ga & 0xffffffffu),                  // global_addr[31:0]
               ((unsigned)((ga >> 32) & 0x01ffffffu)) | (2u << 30) }; // addr[56:32]|type=2
    v8i g1 = { (int)(1u << 16),                               // data_size=1 (2B), mask=0
               (int)(((unsigned)FPAD & 0xffffu) << 16),       // tensor_dim0[15:0] @63:48
               (int)(1u << 16),                               // dim0 hi=0 | tensor_dim1=1
               (int)((unsigned)FPAD << 16),                   // tile_dim0 @127:112
               1,                                             // tile_dim1=1, tile_dim2=0
               (int)FPAD,                                     // tensor_dim0_stride lo32
               0, 0 };
    v4i gz = {0,0,0,0};
#if defined(__clang_major__) && (__clang_major__ >= 23)
    v8i gz8 = {0,0,0,0,0,0,0,0};
    __builtin_amdgcn_tensor_load_to_lds(g0, g1, gz, gz, gz8, 0);
#else
    __builtin_amdgcn_tensor_load_to_lds(g0, g1, gz, gz, 0);
#endif
  }

  // ---- async global->LDS copy of the f32 sample slab (b64 = 8B aligned) ----
  {
    unsigned long long base = (unsigned long long)(size_t)wn;
    for (int u = lane; u < (8*FPAD)/2; u += 32) {
      int bw = u / (FPAD/2);
      int pr = u - bw*(FPAD/2);
      int elem = pr * 2;
      int idx = t0 + elem;
      if (idx > WNLEN - 2) idx = WNLEN - 2;   // clamp: only hits zero-tap region
      unsigned goff = (unsigned)((((size_t)bw*12 + band)*WNLEN + (size_t)idx) * 4u);
      unsigned ldsb = (unsigned)(size_t)(&xf[bw*FPAD + elem]);
      asm volatile("global_load_async_to_lds_b64 %0, %1, %2"
                   :: "v"(ldsb), "v"(goff), "s"(base) : "memory");
    }
    asm volatile("s_wait_asynccnt 0x0" ::: "memory");
  }
  __syncthreads();
  for (int u = lane; u < 8*FPAD; u += 32) xs[u] = (_Float16)xf[u];  // f32 -> f16
  __builtin_amdgcn_s_wait_tensorcnt(0);
  __syncthreads();

  int h  = lane >> 4;        // half-wave selector
  int mn = lane & 15;
  int blkA = mn >> 3, bA = mn & 7;
  const _Float16* xrow = xs + bA*FPAD + 16*blkA;
  const _Float16* hb   = hs + 16 - mn;
  v8f acc = {0.f,0.f,0.f,0.f,0.f,0.f,0.f,0.f};
  for (int kt = 0; kt < 33; ++kt) {
    int k0 = kt * 32;
    v16h av, bv;
#pragma unroll
    for (int v = 0; v < 8; ++v) {
      int ka = (v < 4) ? (2*v + 8*h) : (16 + 2*(v-4) + 8*h);   // A 16x32 f16 layout
      av[2*v]   = xrow[k0 + ka];
      av[2*v+1] = xrow[k0 + ka + 1];
      int kb = ka;                                             // B 32x16 f16 layout
      bv[2*v]   = hb[k0 + kb];
      bv[2*v+1] = hb[k0 + kb + 1];
    }
    acc = __builtin_amdgcn_wmma_f32_16x16x32_f16(false, av, false, bv,
                                                 (short)0, acc, false, false);
  }
  // C layout: VGPR r holds M=r (lanes 0-15) / M=r+8 (lanes 16-31), N = lane%16.
  // M = 8*blk + b  ->  b = r, blk = h.
  int t = t0 + 16*h + mn;
#pragma unroll
  for (int r = 0; r < 8; ++r) {
    int bw = r;                    // wn batch row (= batch*2 + ch)
    int batch = bw >> 1;
    float g   = fxp[batch*25 + band];
    float dec = fxp[batch*25 + 12 + band]*10.f + 1.f;
    float env = __expf(-dec * (float)t * (1.f/65535.f));
    atomicAdd(&ir[(size_t)bw*NSAMP + t], acc[r] * g * env * (1.f/12.f));
  }
}

// ---------------------------------------------------------------------------
// Host launcher
// ---------------------------------------------------------------------------
extern "C" void kernel_launch(void* const* d_in, const int* in_sizes, int n_in,
                              void* d_out, int out_size, void* d_ws, size_t ws_size,
                              hipStream_t stream) {
  (void)in_sizes; (void)n_in; (void)out_size;
  const float* tracks  = (const float*)d_in[0];
  const float* tparams = (const float*)d_in[1];
  const float* fxp     = (const float*)d_in[2];
  const float* mbp     = (const float*)d_in[3];
  const float* wn      = (const float*)d_in[4];
  float* out = (float*)d_out;

  char* w = (char*)d_ws;
  size_t off = 0;
  auto carve = [&](size_t bytes) -> void* {
    void* p = w + off;
    off = (off + bytes + 255) & ~(size_t)255;
    return p;
  };
  SigP*      P      = (SigP*)carve(40 * sizeof(SigP));
  _Float16*  filtH  = (_Float16*)carve((size_t)12 * FPAD * sizeof(_Float16));
  float2*    cA     = (float2*)carve((size_t)8 * NFFT * sizeof(float2));
  float2*    cB     = (float2*)carve((size_t)8 * NFFT * sizeof(float2));
  float*     xG     = (float*)carve((size_t)8 * T_LEN * sizeof(float));
  float*     master = (float*)carve((size_t)8 * T_LEN * sizeof(float));
  float*     bus    = (float*)carve((size_t)8 * T_LEN * sizeof(float));
  float*     ir     = (float*)carve((size_t)8 * NSAMP * sizeof(float));
  if (off > ws_size) return;   // insufficient scratch

  hipMemsetAsync(master, 0, (size_t)8 * T_LEN * sizeof(float), stream);
  hipMemsetAsync(bus,    0, (size_t)8 * T_LEN * sizeof(float), stream);
  hipMemsetAsync(ir,     0, (size_t)8 * NSAMP * sizeof(float), stream);

  k_params<<<dim3(1), dim3(64), 0, stream>>>(P, tparams, mbp);
  k_make_filters<<<dim3(12), dim3(256), 0, stream>>>(filtH);

  const dim3 gF(NFFT/256, 8), gT(T_LEN/256, 8), b256(256);
  auto fwd = [&](float2* buf) {
    k_fft_cols<<<dim3(NROW/4, 8), b256, 0, stream>>>(buf, -1.f, 1, 1.f);
    k_fft_rows<<<dim3(NCOL,   8), b256, 0, stream>>>(buf, -1.f, 0, 1.f);
  };
  auto inv = [&](float2* buf) {
    k_fft_rows<<<dim3(NCOL,   8), b256, 0, stream>>>(buf,  1.f, 1, 1.f);
    k_fft_cols<<<dim3(NROW/4, 8), b256, 0, stream>>>(buf,  1.f, 0, 1.f/(float)NFFT);
  };

  // ---- per-batch track chain: gain -> EQ -> compressor -> pan/send mix ----
  for (int b = 0; b < 4; ++b) {
    k_pack_tracks<<<gF, b256, 0, stream>>>(cA, tracks, P, b);
    fwd(cA);
    k_eq_mul<<<gF, b256, 0, stream>>>(cA, P, b*8);
    inv(cA);
    k_comp_gc<<<gF, b256, 0, stream>>>(cA, xG, P, b*8);
    fwd(cA);
    k_h1p_mul<<<gF, b256, 0, stream>>>(cA, P, b*8);
    inv(cA);
    k_comp_mix<<<gT, b256, 0, stream>>>(cA, xG, master, bus, P, b);
  }

  // ---- reverb: WMMA filterbank -> IR, then FFT convolution with the bus ----
  k_fir_wmma<<<dim3(NSAMP/32, 12), dim3(32), 0, stream>>>(wn, filtH, fxp, ir);
  k_pack_real<<<gF, b256, 0, stream>>>(cA, bus, T_LEN);
  k_pack_real<<<gF, b256, 0, stream>>>(cB, ir, NSAMP);
  fwd(cA);
  fwd(cB);
  k_cmul<<<gF, b256, 0, stream>>>(cA, cB);
  inv(cA);
  k_wet_mix<<<gT, b256, 0, stream>>>(master, bus, cA, fxp);

  // ---- master chain: EQ -> compressor -> output ----
  k_pack_real<<<gF, b256, 0, stream>>>(cA, master, T_LEN);
  fwd(cA);
  k_eq_mul<<<gF, b256, 0, stream>>>(cA, P, 32);
  inv(cA);
  k_comp_gc<<<gF, b256, 0, stream>>>(cA, xG, P, 32);
  fwd(cA);
  k_h1p_mul<<<gF, b256, 0, stream>>>(cA, P, 32);
  inv(cA);
  k_final<<<gT, b256, 0, stream>>>(cA, xG, P, out);
}
